// GraphAttentionLayer_3573412790430
// MI455X (gfx1250) — matrix operand order repaired and checked by reading, not compiled, not run on hardware
//
#include <hip/hip_runtime.h>
#include <hip/hip_bf16.h>

typedef __attribute__((ext_vector_type(16))) __bf16 bf16x16;
typedef __attribute__((ext_vector_type(8)))  float  f32x8;
typedef unsigned int u32;
typedef __attribute__((ext_vector_type(4))) u32 u32x4;
typedef __attribute__((ext_vector_type(8))) u32 u32x8;

#define ALPHA 0.2f
#define F 256            // IN_F == OUT_F
#define NTOK 1024        // entities per batch
#define SLAB_STRIDE 80   // bytes per f-row in LDS slab: 64 data + 16 pad (20-bank lane step)
#define SLAB_BYTES (256 * SLAB_STRIDE)   // 20 KB per buffer

// ---------------------------------------------------------------------------
// TDM: DMA a 2D bf16 tile (tile_d0 contiguous elems x tile_d1 rows) from
// global memory into LDS, with hardware row padding 64B->80B
// (pad_interval=3 -> every 16 dwords, pad_amount=3 -> +4 dwords).
// Issue from ONE wave only (EXEC ignored; uniform SGPR descriptors).
// ---------------------------------------------------------------------------
__device__ __forceinline__ void tdm_load_2d_bf16(unsigned lds_addr, const void* gptr,
                                                 u32 tensor_d0, u32 tensor_d1,
                                                 u32 stride_d0, u32 tile_d0, u32 tile_d1)
{
    unsigned long long ga = (unsigned long long)gptr;
    u32x4 g0;
    g0[0] = 1u;                                   // count=1 (valid user descriptor)
    g0[1] = lds_addr;                             // LDS byte address
    g0[2] = (u32)ga;                              // global_addr[31:0]
    g0[3] = ((u32)(ga >> 32) & 0x01FFFFFFu)       // global_addr[56:32]
          | (2u << 30);                           // type=2 ("image")
    u32x8 g1;
    g1[0] = (1u << 16)                            // data_size = 2 bytes
          | (1u << 20)                            // pad_enable
          | (3u << 22)                            // pad_interval: 16 dwords (64B)
          | (3u << 25);                           // pad_amount: 4 dwords (16B)
    g1[1] = (tensor_d0 & 0xFFFFu) << 16;          // tensor_dim0[15:0]
    g1[2] = (tensor_d0 >> 16) | ((tensor_d1 & 0xFFFFu) << 16);
    g1[3] = (tensor_d1 >> 16) | (tile_d0 << 16);  // tile_dim0
    g1[4] = tile_d1;                              // tile_dim1 ([15:0]); tile_dim2=0
    g1[5] = stride_d0;                            // tensor_dim0_stride[31:0]
    g1[6] = 0;                                    // stride[47:32]=0, dim1_stride=0
    g1[7] = 0;
    asm volatile("tensor_load_to_lds %0, %1" :: "s"(g0), "s"(g1) : "memory");
}

__device__ __forceinline__ unsigned lds_off(const void* p) {
    // flat addr of a __shared__ object: addr[31:0] is the LDS byte offset
    return (unsigned)(unsigned long long)p;
}

// ---------------------------------------------------------------------------
// Kernel 0: WT[f][k] = bf16(W[k][f])   (256x256, one block)
// ---------------------------------------------------------------------------
__global__ void gat_wt_kernel(const float* __restrict__ W, __bf16* __restrict__ WT) {
    const int f = threadIdx.x;                 // 0..255
    #pragma unroll 4
    for (int k = 0; k < F; ++k) {
        WT[f * F + k] = (__bf16)W[k * F + f];  // coalesced read across lanes
    }
}

// ---------------------------------------------------------------------------
// Kernel 1: h = x @ W (bf16 WMMA, f32 acc), store hT[b][f][i] bf16,
//           s1[i] = h[i,:]·a1, s2[i] = h[i,:]·a2
// Grid: ntok/128 blocks x 256 threads (8 waves; wave owns 16 token rows).
// ---------------------------------------------------------------------------
__global__ void __launch_bounds__(256)
gat_h_kernel(const float* __restrict__ x, const __bf16* __restrict__ WT,
             const float* __restrict__ a, __bf16* __restrict__ hT,
             float* __restrict__ s1, float* __restrict__ s2)
{
    __shared__ unsigned char slab[2][SLAB_BYTES];      // double-buffered, 40 KB
    const int tid  = threadIdx.x;
    const int wave = tid >> 5;
    const int lane = tid & 31;
    const int lh   = lane >> 4;        // half-wave select (K-half)
    const int ln   = lane & 15;

    const int tok0  = blockIdx.x * 128;            // global token base of WG
    const int row   = tok0 + wave * 16 + ln;       // this lane's A row (token)
    const int batch = tok0 / NTOK;                 // 128 | 1024, WG is batch-pure
    const int iloc0 = (tok0 % NTOK) + wave * 16;   // local token base of wave

    f32x8 acc[16];
    #pragma unroll
    for (int t = 0; t < 16; ++t) acc[t] = (f32x8)0.0f;

    // prologue: DMA WT[:, 0:32] into buffer 0
    if (wave == 0)
        tdm_load_2d_bf16(lds_off(&slab[0][0]), WT, F, F, F, 32, F);

    for (int k0 = 0; k0 < F; k0 += 32) {
        const int cur = (k0 >> 5) & 1;

        // A tile: x[row, k0 + {16lh..16lh+15}] — independent of LDS,
        // issue before the barriers so the loads overlap the TDM wait.
        float xv[16] __attribute__((aligned(16)));
        {
            const float* xp = x + row * F + k0 + 16 * lh;
            ((float4*)xv)[0] = ((const float4*)xp)[0];
            ((float4*)xv)[1] = ((const float4*)xp)[1];
            ((float4*)xv)[2] = ((const float4*)xp)[2];
            ((float4*)xv)[3] = ((const float4*)xp)[3];
        }

        __syncthreads();   // everyone done reading the buffer we are about to refill
        if (wave == 0) {
            if (k0 + 32 < F) {
                tdm_load_2d_bf16(lds_off(&slab[cur ^ 1][0]), WT + (k0 + 32), F, F, F, 32, F);
                __builtin_amdgcn_s_wait_tensorcnt(1);   // current tile done (in-order)
            } else {
                __builtin_amdgcn_s_wait_tensorcnt(0);
            }
        }
        __syncthreads();   // publish current tile to all waves

        bf16x16 A;
        #pragma unroll
        for (int e = 0; e < 16; ++e) A[e] = (__bf16)xv[e];

        const unsigned char* sb = &slab[cur][0];
        #pragma unroll
        for (int t = 0; t < 16; ++t) {
            const uint4* bp = (const uint4*)(sb + (t * 16 + ln) * SLAB_STRIDE + lh * 32);
            union { uint4 u[2]; bf16x16 v; } B;
            B.u[0] = bp[0]; B.u[1] = bp[1];
            acc[t] = __builtin_amdgcn_wmma_f32_16x16x32_bf16(
                false, A, false, B.v, (short)0, acc[t], false, false);
        }
    }

    // Epilogue: store hT (transposed, bf16) + accumulate s1/s2 partials
    f32x8 sv1 = (f32x8)0.0f, sv2 = (f32x8)0.0f;
    #pragma unroll
    for (int t = 0; t < 16; ++t) {
        const int f = t * 16 + ln;
        const float a1v = a[f];
        const float a2v = a[F + f];
        union { __bf16 h[8]; uint4 u; } P;
        #pragma unroll
        for (int r = 0; r < 8; ++r) {
            const float v = acc[t][r];
            P.h[r] = (__bf16)v;
            sv1[r] += v * a1v;
            sv2[r] += v * a2v;
        }
        // rows r=0..7 are tokens iloc0+8*lh+r (contiguous) -> one b128 store
        *(uint4*)(hT + (size_t)(batch * F + f) * NTOK + iloc0 + 8 * lh) = P.u;
    }
    // reduce across the 16-lane half group (f dimension)
    #pragma unroll
    for (int off = 1; off < 16; off <<= 1) {
        #pragma unroll
        for (int r = 0; r < 8; ++r) {
            sv1[r] += __shfl_xor(sv1[r], off, 32);
            sv2[r] += __shfl_xor(sv2[r], off, 32);
        }
    }
    if (ln == 0) {
        const int base = tok0 + wave * 16 + 8 * lh;
        #pragma unroll
        for (int r = 0; r < 8; ++r) {
            s1[base + r] = sv1[r];
            s2[base + r] = sv2[r];
        }
    }
}

// ---------------------------------------------------------------------------
// Kernel 2: fused leaky-relu + online softmax + (attention @ h) via WMMA.
// Grid: B * (N/128) blocks x 256 threads. Wave owns 16 i-rows x all 256 f.
// ---------------------------------------------------------------------------
__global__ void __launch_bounds__(256)
gat_attn_kernel(const __bf16* __restrict__ hT, const float* __restrict__ s1g,
                const float* __restrict__ s2g, float* __restrict__ out)
{
    __shared__ unsigned char slab[2][SLAB_BYTES];      // double-buffered, 40 KB
    const int tid  = threadIdx.x;
    const int wave = tid >> 5;
    const int lane = tid & 31;
    const int lh   = lane >> 4;
    const int ln   = lane & 15;

    const int batch = blockIdx.x >> 3;            // 8 i-tiles per batch
    const int itile = blockIdx.x & 7;
    const int i0    = itile * 128 + wave * 16;    // local i base for this wave
    const int gtok  = batch * NTOK + i0;

    const float s1v = s1g[gtok + ln];             // this lane's row score (row = ln)
    const __bf16* hTb = hT + (size_t)batch * F * NTOK;

    float m = -3.0e38f, lsum = 0.0f;
    f32x8 acc[16];
    #pragma unroll
    for (int t = 0; t < 16; ++t) acc[t] = (f32x8)0.0f;

    // prologue: DMA hT[:, 0:32] into buffer 0
    if (wave == 0)
        tdm_load_2d_bf16(lds_off(&slab[0][0]), hTb, NTOK, F, NTOK, 32, F);

    for (int j0 = 0; j0 < NTOK; j0 += 32) {
        const int cur = (j0 >> 5) & 1;

        // scores: s2 chunk for this lane's K positions (independent of LDS)
        float ev[16] __attribute__((aligned(16)));
        {
            const float* sp = s2g + batch * NTOK + j0 + 16 * lh;
            ((float4*)ev)[0] = ((const float4*)sp)[0];
            ((float4*)ev)[1] = ((const float4*)sp)[1];
            ((float4*)ev)[2] = ((const float4*)sp)[2];
            ((float4*)ev)[3] = ((const float4*)sp)[3];
        }

        __syncthreads();   // all waves done reading the buffer being refilled
        if (wave == 0) {
            if (j0 + 32 < NTOK) {
                tdm_load_2d_bf16(lds_off(&slab[cur ^ 1][0]), hTb + (j0 + 32),
                                 NTOK, F, NTOK, 32, F);
                __builtin_amdgcn_s_wait_tensorcnt(1);
            } else {
                __builtin_amdgcn_s_wait_tensorcnt(0);
            }
        }
        __syncthreads();   // publish current tile

        // e[i, j0+k] = leakyrelu(s1[i] + s2[j0+k]) in A-register layout
        float tmax = -3.0e38f;
        #pragma unroll
        for (int e = 0; e < 16; ++e) {
            float v = s1v + ev[e];
            v = v > 0.0f ? v : ALPHA * v;
            ev[e] = v;
            tmax = fmaxf(tmax, v);
        }
        tmax = fmaxf(tmax, __shfl_xor(tmax, 16, 32));   // combine K-halves
        const float mnew  = fmaxf(m, tmax);
        const float scale = __expf(m - mnew);

        bf16x16 A;
        float psum = 0.0f;
        #pragma unroll
        for (int e = 0; e < 16; ++e) {
            const float p = __expf(ev[e] - mnew);
            psum += p;
            A[e] = (__bf16)p;
        }
        psum += __shfl_xor(psum, 16, 32);
        lsum = lsum * scale + psum;
        m = mnew;

        // rescale accumulators: acc VGPR r holds row (r + 8*lh); stats live in lane=row
        float scr[8];
        #pragma unroll
        for (int r = 0; r < 8; ++r) scr[r] = __shfl(scale, r + 8 * lh, 32);
        #pragma unroll
        for (int t = 0; t < 16; ++t) {
            #pragma unroll
            for (int r = 0; r < 8; ++r) acc[t][r] *= scr[r];
        }

        // acc[t] += P(16x32) x h[j0:j0+32, 16t:16t+16]
        const unsigned char* sb = &slab[cur][0];
        #pragma unroll
        for (int t = 0; t < 16; ++t) {
            const uint4* bp = (const uint4*)(sb + (t * 16 + ln) * SLAB_STRIDE + lh * 32);
            union { uint4 u[2]; bf16x16 v; } B;
            B.u[0] = bp[0]; B.u[1] = bp[1];
            acc[t] = __builtin_amdgcn_wmma_f32_16x16x32_bf16(
                false, A, false, B.v, (short)0, acc[t], false, false);
        }
    }

    // out[b, i, f] = acc / l  (per-row 1/l broadcast to D layout)
    float invl[8];
    #pragma unroll
    for (int r = 0; r < 8; ++r) invl[r] = 1.0f / __shfl(lsum, r + 8 * lh, 32);
    #pragma unroll
    for (int t = 0; t < 16; ++t) {
        const int f = t * 16 + ln;
        #pragma unroll
        for (int r = 0; r < 8; ++r) {
            out[(size_t)(batch * NTOK + i0 + r + 8 * lh) * F + f] = acc[t][r] * invl[r];
        }
    }
}

// ---------------------------------------------------------------------------
extern "C" void kernel_launch(void* const* d_in, const int* in_sizes, int n_in,
                              void* d_out, int out_size, void* d_ws, size_t ws_size,
                              hipStream_t stream) {
    (void)n_in; (void)out_size; (void)ws_size;
    const float* x = (const float*)d_in[0];   // (B, N, 256) f32
    const float* W = (const float*)d_in[1];   // (256, 256)  f32
    const float* a = (const float*)d_in[2];   // (512, 1)    f32

    const int ntok = in_sizes[0] / F;         // B*N = 65536
    const int nbat = ntok / NTOK;             // 64

    // workspace layout: hT bf16 | WT bf16 | s1 f32 | s2 f32
    char* ws = (char*)d_ws;
    __bf16* hT = (__bf16*)ws;
    size_t off = (size_t)ntok * F * sizeof(__bf16);
    __bf16* WT = (__bf16*)(ws + off);
    off += (size_t)F * F * sizeof(__bf16);
    float* s1 = (float*)(ws + off);
    off += (size_t)ntok * sizeof(float);
    float* s2 = (float*)(ws + off);

    gat_wt_kernel<<<1, 256, 0, stream>>>(W, WT);
    gat_h_kernel<<<ntok / 128, 256, 0, stream>>>(x, WT, a, hT, s1, s2);
    gat_attn_kernel<<<nbat * (NTOK / 128), 256, 0, stream>>>(hT, s1, s2, (float*)d_out);
}
